// LocationAwareAttention_57939108823195
// MI455X (gfx1250) — matrix operand order repaired and checked by reading, not compiled, not run on hardware
//
#include <hip/hip_runtime.h>
#include <hip/hip_bf16.h>
#include <math.h>
#include <string.h>

// Problem constants (match reference)
#define B_    16
#define T_    4096
#define D_    512
#define C_    10
#define FW_   100
#define KW_   (2*FW_+1)     // 201
#define SHARP_ 2.0f

// GEMM geometry: K augmented with conv channels + zero pad
#define KTOT  544           // 512 value + 10 conv + 22 pad  (= 17 * 32)
#define KCH   32
#define NCHUNK (KTOT/KCH)   // 17
#define BM    128           // M rows per workgroup (full N=512 per workgroup)
#define GEMM_THREADS 512    // 16 waves (wave32)
#define LDS_PITCH 80        // 64 data bytes (32 bf16) + 16 pad -> conflict-free b128
#define ABYTES (BM * LDS_PITCH)   // 10240
#define BBYTES (D_ * LDS_PITCH)   // 40960

#define SEG   32            // context reduction segments over T

typedef __attribute__((ext_vector_type(8)))  float   v8f;
typedef __attribute__((ext_vector_type(16))) __bf16  v16bf;
typedef __attribute__((ext_vector_type(4)))  int     i4;
typedef __attribute__((address_space(1))) i4* gas_i4p;   // global int4*
typedef __attribute__((address_space(3))) i4* las_i4p;   // LDS int4*

// ---------- helpers ----------
static __device__ __forceinline__ unsigned pack_bf2(float x, float y) {
    __bf16 a = (__bf16)x, b = (__bf16)y;
    unsigned short ua, ub;
    __builtin_memcpy(&ua, &a, 2);
    __builtin_memcpy(&ub, &b, 2);
    return (unsigned)ua | ((unsigned)ub << 16);
}

static __device__ __forceinline__ v16bf load_frag_pair(const char* p0, const char* p1) {
    union { uint4 q[2]; v16bf v; } u;
    u.q[0] = *(const uint4*)p0;
    u.q[1] = *(const uint4*)p1;
    return u.v;
}

// CDNA5 async global->LDS copy (ASYNCcnt-tracked, no VGPR round trip)
static __device__ __forceinline__ void async_b128(const void* g, void* l) {
    __builtin_amdgcn_global_load_async_to_lds_b128((gas_i4p)g, (las_i4p)l, 0, 0);
}

static __device__ __forceinline__ void wait_async0() {
    asm volatile("s_wait_asynccnt 0x0" ::: "memory");
}

// ---------- kernel 0a: fused B matrix, CHUNK-MAJOR [17][512][32] bf16 ----------
// logical column k of row e:  k<512 -> W_enc[e][k]; 512..521 -> W_att[e][k-512]; else 0
__global__ __launch_bounds__(256)
void laa_buildB_kernel(const float* __restrict__ W_enc,
                       const float* __restrict__ W_att,
                       __bf16* __restrict__ Bw) {
    int o = blockIdx.x * 256 + threadIdx.x;
    if (o >= D_ * KTOT) return;
    int e = o / KTOT, k = o % KTOT;
    float v;
    if (k < D_)            v = W_enc[e * D_ + k];
    else if (k < D_ + C_)  v = W_att[e * C_ + (k - D_)];
    else                   v = 0.0f;
    int kc = k / KCH, kj = k % KCH;
    Bw[((size_t)kc * D_ + e) * KCH + kj] = (__bf16)v;
}

// ---------- kernel 0b: location conv -> padded bf16 A-extension [B*T][32] ----------
__global__ __launch_bounds__(256)
void laa_conv_kernel(const float* __restrict__ att_prev,  // [B][T]
                     const float* __restrict__ W_conv,    // [C][1][KW]
                     __bf16* __restrict__ convP) {        // [B*T][32]
    __shared__ float wc[C_ * KW_];
    for (int i = threadIdx.x; i < C_ * KW_; i += 256) wc[i] = W_conv[i];
    __syncthreads();

    int o = blockIdx.x * 256 + threadIdx.x;   // 0..B*T-1 (grid sized exactly)
    int b = o / T_, t = o % T_;
    const float* ap = att_prev + (size_t)b * T_;

    float acc[C_];
    #pragma unroll
    for (int c = 0; c < C_; c++) acc[c] = 0.0f;

    for (int j = 0; j < KW_; j++) {
        int ts = t + j - FW_;
        if (ts < 0 || ts >= T_) continue;     // zero padding (same as lax conv)
        float a = ap[ts];
        #pragma unroll
        for (int c = 0; c < C_; c++) acc[c] += a * wc[c * KW_ + j];
    }
    __bf16* dst = convP + (size_t)o * KCH;
    #pragma unroll
    for (int c = 0; c < C_; c++)  dst[c] = (__bf16)acc[c];
    #pragma unroll
    for (int c = C_; c < KCH; c++) dst[c] = (__bf16)0.0f;
}

// ---------- kernel 0c: bias2[b][e] = query[b]·W_dec[e] + b_enc[e] ----------
__global__ __launch_bounds__(256)
void laa_bias_kernel(const float* __restrict__ query,
                     const float* __restrict__ W_dec,
                     const float* __restrict__ b_enc,
                     float* __restrict__ bias2) {
    int o = blockIdx.x * 256 + threadIdx.x;   // 0..B*D-1
    int b = o >> 9, e = o & (D_ - 1);
    const float* q = query + (size_t)b * D_;
    const float* w = W_dec + (size_t)e * D_;
    float acc = 0.0f;
    for (int d = 0; d < D_; d++) acc += q[d] * w[d];
    bias2[o] = acc + b_enc[e];
}

// ---------- GEMM staging: chunk kc -> LDS buffers (async B, cvt A) ----------
static __device__ __forceinline__ void stage_chunk(
    int kc, char* Abuf, char* Bbuf,
    const float*  __restrict__ value,
    const __bf16* __restrict__ Bw,
    const __bf16* __restrict__ convP,
    int bm0, int tid)
{
    // B tile: contiguous 32 KB (chunk-major layout) -> lane-consecutive async b128
    const char* bsrc = (const char*)Bw + (size_t)kc * (size_t)(D_ * KCH * 2);
    #pragma unroll
    for (int it = 0; it < 4; it++) {
        int i = it * GEMM_THREADS + tid;           // 0..2047, 16B each
        async_b128(bsrc + (size_t)i * 16, Bbuf + (i >> 2) * LDS_PITCH + (i & 3) * 16);
    }
    if (kc < 16) {
        // A tile: value f32 -> bf16 (needs VALU cvt, so regular loads)
        float4 va[2];
        #pragma unroll
        for (int it = 0; it < 2; it++) {
            int idx = it * GEMM_THREADS + tid;     // 0..1023
            int row = idx >> 3, q = idx & 7;
            va[it] = *(const float4*)(value + (size_t)(bm0 + row) * D_ + kc * KCH + q * 4);
        }
        #pragma unroll
        for (int it = 0; it < 2; it++) {
            int idx = it * GEMM_THREADS + tid;
            int row = idx >> 3, q = idx & 7;
            uint2 p;
            p.x = pack_bf2(va[it].x, va[it].y);
            p.y = pack_bf2(va[it].z, va[it].w);
            *(uint2*)(Abuf + row * LDS_PITCH + q * 8) = p;
        }
    } else {
        // conv extension chunk: contiguous 8 KB of bf16 -> one async b128 per thread
        const char* csrc = (const char*)convP + (size_t)bm0 * (KCH * 2);
        async_b128(csrc + (size_t)tid * 16, Abuf + (tid >> 2) * LDS_PITCH + (tid & 3) * 16);
    }
}

// ---------- kernel 1: WMMA GEMM, double-buffered LDS, fused tanh/w_g epilogue ----------
__global__ __launch_bounds__(GEMM_THREADS)
void laa_gemm_kernel(const float*  __restrict__ value,   // [B*T][512] f32
                     const __bf16* __restrict__ Bw,      // [17][512][32] bf16
                     const __bf16* __restrict__ convP,   // [B*T][32]  bf16
                     const float*  __restrict__ bias2,   // [B][512]
                     const float*  __restrict__ w_g,     // [512]
                     const float*  __restrict__ b_g,     // [1]
                     const int*    __restrict__ lens,    // [B]
                     float* __restrict__ e_out) {        // [B*T] (sharpened, masked)
    __shared__ __attribute__((aligned(16))) char As[2 * ABYTES];   // 20 KB
    __shared__ __attribute__((aligned(16))) char Bs[2 * BBYTES];   // 80 KB
    __shared__ float e_buf[BM];

    const int tid  = threadIdx.x;
    const int lane = tid & 31;
    const int wid  = tid >> 5;         // 0..15
    const int wm   = wid >> 3;         // 0..1  (M group of 64 rows)
    const int wn   = wid & 7;          // 0..7  (N group of 64 cols)
    const int bm0  = blockIdx.x * BM;  // global M row base
    const int b    = bm0 / T_;         // batch (BM divides T)

    if (tid < BM) e_buf[tid] = 0.0f;

    v8f acc[4][4] = {};

    // per-lane N-column constants (4 columns per lane)
    float bias_n[4], wg_n[4];
    #pragma unroll
    for (int ni = 0; ni < 4; ni++) {
        int gn = wn * 64 + ni * 16 + (lane & 15);
        bias_n[ni] = bias2[b * D_ + gn];
        wg_n[ni]   = w_g[gn];
    }

    // prologue: stage chunk 0 into buffer 0
    stage_chunk(0, As, Bs, value, Bw, convP, bm0, tid);
    wait_async0();
    __syncthreads();

    for (int kc = 0; kc < NCHUNK; kc++) {
        const int cur = kc & 1;
        char* Abuf = As + cur * ABYTES;
        char* Bbuf = Bs + cur * BBYTES;

        // overlap: stage next chunk into the other buffer while we compute
        if (kc + 1 < NCHUNK)
            stage_chunk(kc + 1, As + (cur ^ 1) * ABYTES, Bs + (cur ^ 1) * BBYTES,
                        value, Bw, convP, bm0, tid);

        // ---- B fragments: lanes 0-15 -> K 0..15 (bytes 0..31), lanes 16-31 -> K 16..31 ----
        const int bo0 = (lane < 16) ? 0 : 32;
        v16bf bfrag[4];
        #pragma unroll
        for (int ni = 0; ni < 4; ni++) {
            const char* base = Bbuf + (wn * 64 + ni * 16 + (lane & 15)) * LDS_PITCH;
            bfrag[ni] = load_frag_pair(base + bo0, base + bo0 + 16);
        }
        // ---- A fragments + WMMA: lanes 0-15 -> K 0..15 (bytes 0..31), lanes 16-31 -> K 16..31 ----
        const int ao0 = (lane < 16) ? 0 : 32;
        #pragma unroll
        for (int mi = 0; mi < 4; mi++) {
            const char* base = Abuf + (wm * 64 + mi * 16 + (lane & 15)) * LDS_PITCH;
            v16bf afrag = load_frag_pair(base + ao0, base + ao0 + 16);
            #pragma unroll
            for (int ni = 0; ni < 4; ni++) {
                acc[mi][ni] = __builtin_amdgcn_wmma_f32_16x16x32_bf16(
                    false, afrag, false, bfrag[ni], (short)0, acc[mi][ni], false, false);
            }
        }

        if (kc + 1 < NCHUNK) wait_async0();   // staging of next buffer complete
        __syncthreads();
    }

    // ---- epilogue: e[t] = sum_n w_g[n] * tanh(h[t,n]) ----
    #pragma unroll
    for (int mi = 0; mi < 4; mi++) {
        #pragma unroll
        for (int r = 0; r < 8; r++) {
            float v = 0.0f;
            #pragma unroll
            for (int ni = 0; ni < 4; ni++)
                v += wg_n[ni] * tanhf(acc[mi][ni][r] + bias_n[ni]);
            // reduce across the 16 lanes sharing this row (different n)
            v += __shfl_xor(v, 1, 32);
            v += __shfl_xor(v, 2, 32);
            v += __shfl_xor(v, 4, 32);
            v += __shfl_xor(v, 8, 32);
            if ((lane & 15) == 0) {
                int tl = wm * 64 + mi * 16 + r + ((lane < 16) ? 0 : 8);
                atomicAdd(&e_buf[tl], v);   // sum across the 8 N-waves
            }
        }
    }
    __syncthreads();
    if (tid < BM) {
        int tloc = (bm0 % T_) + tid;
        float e = e_buf[tid] + b_g[0];
        e_out[bm0 + tid] = (tloc >= lens[b]) ? -2e30f : SHARP_ * e;
    }
}

// ---------- kernel 2: softmax over T per batch ----------
__global__ __launch_bounds__(256)
void laa_softmax_kernel(const float* __restrict__ e, float* __restrict__ att) {
    __shared__ float red[256];
    int b = blockIdx.x;
    const float* eb = e + (size_t)b * T_;

    float m = -3.4e38f;
    for (int t = threadIdx.x; t < T_; t += 256) m = fmaxf(m, eb[t]);
    red[threadIdx.x] = m; __syncthreads();
    for (int s = 128; s > 0; s >>= 1) {
        if (threadIdx.x < s) red[threadIdx.x] = fmaxf(red[threadIdx.x], red[threadIdx.x + s]);
        __syncthreads();
    }
    float mx = red[0]; __syncthreads();

    float sum = 0.0f;
    for (int t = threadIdx.x; t < T_; t += 256) sum += expf(eb[t] - mx);
    red[threadIdx.x] = sum; __syncthreads();
    for (int s = 128; s > 0; s >>= 1) {
        if (threadIdx.x < s) red[threadIdx.x] += red[threadIdx.x + s];
        __syncthreads();
    }
    float inv = 1.0f / red[0];
    for (int t = threadIdx.x; t < T_; t += 256)
        att[(size_t)b * T_ + t] = expf(eb[t] - mx) * inv;
}

// ---------- kernel 3a/3b: context[b][d] = sum_t value[b,t,d] * att[b,t] ----------
__global__ __launch_bounds__(512)
void laa_ctx_part_kernel(const float* __restrict__ value,
                         const float* __restrict__ att,
                         float* __restrict__ partial) {  // [B][SEG][D]
    int b = blockIdx.x / SEG, s = blockIdx.x % SEG;
    int d = threadIdx.x;
    int t0 = s * (T_ / SEG);
    float acc = 0.0f;
    for (int i = 0; i < T_ / SEG; i++) {
        int t = t0 + i;
        acc += value[((size_t)b * T_ + t) * D_ + d] * att[(size_t)b * T_ + t];
    }
    partial[((size_t)b * SEG + s) * D_ + d] = acc;
}

__global__ __launch_bounds__(256)
void laa_ctx_sum_kernel(const float* __restrict__ partial, float* __restrict__ ctx) {
    int o = blockIdx.x * 256 + threadIdx.x;   // 0..B*D-1
    int b = o >> 9, d = o & (D_ - 1);
    float acc = 0.0f;
    #pragma unroll
    for (int s = 0; s < SEG; s++) acc += partial[((size_t)b * SEG + s) * D_ + d];
    ctx[o] = acc;
}

// ---------- launch ----------
extern "C" void kernel_launch(void* const* d_in, const int* in_sizes, int n_in,
                              void* d_out, int out_size, void* d_ws, size_t ws_size,
                              hipStream_t stream) {
    const float* value    = (const float*)d_in[0];
    const float* query    = (const float*)d_in[1];
    const int*   lens     = (const int*)  d_in[2];
    const float* att_prev = (const float*)d_in[3];
    const float* W_enc    = (const float*)d_in[4];
    const float* b_enc    = (const float*)d_in[5];
    const float* W_dec    = (const float*)d_in[6];
    const float* W_att    = (const float*)d_in[7];
    const float* W_conv   = (const float*)d_in[8];
    const float* w_g      = (const float*)d_in[9];
    const float* b_g      = (const float*)d_in[10];

    char* ws = (char*)d_ws;
    __bf16* Bw      = (__bf16*)(ws);                 // 17*512*32*2     = 557,056 B
    __bf16* convP   = (__bf16*)(ws + 557056);        // 65536*32*2      = 4,194,304 B
    float*  bias2   = (float*) (ws + 4751360);       // 16*512*4        = 32,768 B
    float*  e_ws    = (float*) (ws + 4784128);       // 16*4096*4       = 262,144 B
    float*  partial = (float*) (ws + 5046272);       // 16*32*512*4     = 1,048,576 B
    // total workspace used: ~6.1 MB

    float* ctx = (float*)d_out;               // [B][D]
    float* att = (float*)d_out + B_ * D_;     // [B][T]

    laa_buildB_kernel<<<(D_ * KTOT + 255) / 256, 256, 0, stream>>>(W_enc, W_att, Bw);
    laa_conv_kernel<<<(B_ * T_) / 256, 256, 0, stream>>>(att_prev, W_conv, convP);
    laa_bias_kernel<<<(B_ * D_) / 256, 256, 0, stream>>>(query, W_dec, b_enc, bias2);
    laa_gemm_kernel<<<(B_ * T_) / BM, GEMM_THREADS, 0, stream>>>(
        value, Bw, convP, bias2, w_g, b_g, lens, e_ws);
    laa_softmax_kernel<<<B_, 256, 0, stream>>>(e_ws, att);
    laa_ctx_part_kernel<<<B_ * SEG, 512, 0, stream>>>(value, att, partial);
    laa_ctx_sum_kernel<<<(B_ * D_) / 256, 256, 0, stream>>>(partial, ctx);
}
